// RelationDecoder_51041391345811
// MI455X (gfx1250) — compile-verified, matrix operand order checked by reading, not run on hardware
//
#include <hip/hip_runtime.h>

#define DFEAT 256
#define HID   256
#define K2    512   // 2*DFEAT
#define MT    4     // M-tiles (of 16 edges) per block
#define MB    (16 * MT)   // 64 edges per block

typedef __attribute__((ext_vector_type(16))) __bf16        v16bf;
typedef __attribute__((ext_vector_type(8)))  float         v8f;
typedef __attribute__((ext_vector_type(4)))  unsigned int  uint4v;

union FragU { uint4v u[2]; v16bf v; };

__device__ __forceinline__ unsigned short f2bf(float f) {
    unsigned int u = __float_as_uint(f);
    u += 0x7FFFu + ((u >> 16) & 1u);   // round-to-nearest-even
    return (unsigned short)(u >> 16);
}

// W1 [512,256] f32 (row-major, K x N)  ->  W1t [256,512] bf16 (N x K, contiguous K)
__global__ void w1_transpose_bf16(const float* __restrict__ W1,
                                  unsigned short* __restrict__ W1t) {
    int k = blockIdx.x;    // 0..511
    int n = threadIdx.x;   // 0..255
    W1t[n * K2 + k] = f2bf(W1[k * HID + n]);
}

__global__ __launch_bounds__(512)
void edge_mlp_score(const float* __restrict__ x_src,
                    const float* __restrict__ x_dst,
                    const unsigned short* __restrict__ W1t,
                    const float* __restrict__ b1,
                    const float* __restrict__ W2,
                    const float* __restrict__ b2,
                    const int* __restrict__ ei,    // [2,E] flat
                    const int* __restrict__ nei,   // [2,E] flat
                    float* __restrict__ out,       // [2E]: pos then neg
                    int E)
{
    __shared__ __align__(16) unsigned short sA[MB * K2];  // 64 edges x 512 bf16 = 64 KB
    __shared__ int   sSrc[MB];
    __shared__ int   sDst[MB];
    __shared__ float sScore[MB];

    const int  tid      = threadIdx.x;
    const int  lane     = tid & 31;
    const int  wave     = tid >> 5;                 // 0..15
    const long tileBase = (long)blockIdx.x * MB;
    const long total    = 2L * (long)E;

    // ---- Phase 1: edge indices for this 64-edge tile ----
    if (tid < MB) {
        long r = tileBase + tid;
        int s = 0, d = 0;
        if (r < (long)E)      { s = ei[r];             d = ei[E + r]; }
        else if (r < total)   { long rr = r - E; s = nei[rr]; d = nei[E + rr]; }
        sSrc[tid] = s; sDst[tid] = d;
        sScore[tid] = 0.0f;
    }
    __syncthreads();

    // ---- Phase 2: gather 64 rows x 512 f32 -> bf16 into LDS ----
    #pragma unroll
    for (int it = 0; it < MT; ++it) {
        int base = (tid + it * 512) * 16;   // 16 consecutive elements per chunk
        int e    = base >> 9;               // edge within tile (0..63)
        int pos  = base & 511;              // position within concat(src,dst)
        const float* srcp;
        if (pos < DFEAT) srcp = x_src + (long)sSrc[e] * DFEAT + pos;
        else             srcp = x_dst + (long)sDst[e] * DFEAT + (pos - DFEAT);
        const float4* sp = (const float4*)srcp;
        unsigned int pk[8];
        #pragma unroll
        for (int q = 0; q < 4; ++q) {
            float4 f = sp[q];
            pk[q*2+0] = (unsigned)f2bf(f.x) | ((unsigned)f2bf(f.y) << 16);
            pk[q*2+1] = (unsigned)f2bf(f.z) | ((unsigned)f2bf(f.w) << 16);
        }
        uint4v* dst = (uint4v*)(sA + base);
        dst[0] = (uint4v){pk[0], pk[1], pk[2], pk[3]};
        dst[1] = (uint4v){pk[4], pk[5], pk[6], pk[7]};
    }
    __syncthreads();

    // ---- Phase 3: wave 'wave' computes 64 edges x 16 hidden cols, K=512 ----
    // A (16x32 bf16): lane L<16 -> row M=L, K base +0 ; lane>=16 -> M=L-16, K base +8
    // B (32x16 bf16): lane L<16 -> col N=L, K base +0 ; lane>=16 -> N=L-16, K base +8
    // B fragment reused by MT WMMAs; fragments double-buffered so iteration
    // ko+32's loads are in flight while iteration ko's WMMAs execute.
    const int nCol = (wave << 4) | (lane & 15);
    const int mRow = lane & 15;
    const int koff = (lane >= 16) ? 8 : 0;
    const unsigned short* wcol = W1t + (long)nCol * K2;

    v8f c[MT] = {{}, {}, {}, {}};

    FragU bq, aq[MT];
    bq.u[0] = *(const uint4v*)(wcol + koff);
    bq.u[1] = *(const uint4v*)(wcol + 16 + koff);
    #pragma unroll
    for (int mt = 0; mt < MT; ++mt) {
        const unsigned short* arow = sA + (mt * 16 + mRow) * K2;
        aq[mt].u[0] = *(const uint4v*)(arow + koff);
        aq[mt].u[1] = *(const uint4v*)(arow + 16 + koff);
    }

    #pragma unroll
    for (int ko = 0; ko < K2; ko += 32) {
        const int kn = ko + 32;
        FragU bn, an[MT];
        if (kn < K2) {                       // prefetch next stage
            bn.u[0] = *(const uint4v*)(wcol + kn + koff);
            bn.u[1] = *(const uint4v*)(wcol + kn + 16 + koff);
            #pragma unroll
            for (int mt = 0; mt < MT; ++mt) {
                const unsigned short* arow = sA + (mt * 16 + mRow) * K2;
                an[mt].u[0] = *(const uint4v*)(arow + kn + koff);
                an[mt].u[1] = *(const uint4v*)(arow + kn + 16 + koff);
            }
        }
        #pragma unroll
        for (int mt = 0; mt < MT; ++mt) {
            c[mt] = __builtin_amdgcn_wmma_f32_16x16x32_bf16(
                        /*neg_a=*/false, aq[mt].v, /*neg_b=*/false, bq.v,
                        /*c_mod=*/(short)0, c[mt],
                        /*reuse_a=*/false, /*reuse_b=*/false);
        }
        if (kn < K2) {
            bq = bn;
            #pragma unroll
            for (int mt = 0; mt < MT; ++mt) aq[mt] = an[mt];
        }
    }

    // ---- Epilogue: bias + ReLU + dot with W2, reduce over hidden dim ----
    // C layout: VGPR i: lanes 0-15 -> (M=i, N=lane), lanes16-31 -> (M=8+i, N=lane-16)
    float bias = b1[nCol];
    float w2v  = W2[nCol];
    #pragma unroll
    for (int mt = 0; mt < MT; ++mt) {
        float p[8];
        #pragma unroll
        for (int i = 0; i < 8; ++i) {
            float h = c[mt][i] + bias;
            h = fmaxf(h, 0.0f);
            p[i] = h * w2v;
        }
        // reduce across the 16 lanes of each half-wave (xor<=8 stays in-half)
        #pragma unroll
        for (int i = 0; i < 8; ++i) {
            #pragma unroll
            for (int off = 8; off >= 1; off >>= 1)
                p[i] += __shfl_xor(p[i], off, 32);
        }
        if ((lane & 15) == 0) {
            int mbase = mt * 16 + (lane >> 4) * 8;  // lane 0 -> m 0..7, lane 16 -> 8..15
            #pragma unroll
            for (int i = 0; i < 8; ++i)
                atomicAdd(&sScore[mbase + i], p[i]);   // ds_add_f32
        }
    }
    __syncthreads();

    // ---- Output ----
    if (tid < MB) {
        long r = tileBase + tid;
        if (r < total) out[r] = sScore[tid] + b2[0];
    }
}

extern "C" void kernel_launch(void* const* d_in, const int* in_sizes, int n_in,
                              void* d_out, int out_size, void* d_ws, size_t ws_size,
                              hipStream_t stream) {
    const float* x_src = (const float*)d_in[0];
    const float* x_dst = (const float*)d_in[1];
    const float* W1    = (const float*)d_in[2];
    const float* b1    = (const float*)d_in[3];
    const float* W2    = (const float*)d_in[4];
    const float* b2    = (const float*)d_in[5];
    const int*   ei    = (const int*)d_in[6];
    const int*   nei   = (const int*)d_in[7];
    float*       out   = (float*)d_out;

    int E = in_sizes[6] / 2;                       // edge_index is [2,E]
    unsigned short* W1t = (unsigned short*)d_ws;   // 256x512 bf16 = 256 KB

    hipLaunchKernelGGL(w1_transpose_bf16, dim3(K2), dim3(HID), 0, stream, W1, W1t);

    long total  = 2L * (long)E;
    int  nTiles = (int)((total + MB - 1) / MB);
    hipLaunchKernelGGL(edge_mlp_score, dim3(nTiles), dim3(512), 0, stream,
                       x_src, x_dst, W1t, b1, W2, b2, ei, nei, out, E);
}
